// AttentionLayer_71511205479000
// MI455X (gfx1250) — matrix operand order repaired and checked by reading, not compiled, not run on hardware
//
#include <hip/hip_runtime.h>

// ---------------------------------------------------------------------------
// Attention layer for MI455X (gfx1250, wave32). All matmuls via
// v_wmma_f32_16x16x32_f16; all global->LDS staging except V via the CDNA5 async path
// (global_load_async_to_lds_b128 + s_wait_asynccnt) with double-buffered LDS.
// B=8, S=2048, D=K=V=1024  (~240 GFLOP, compute-bound).
// ---------------------------------------------------------------------------

#define B_  8
#define S_  2048
#define D_  1024
#define M_  (B_ * S_)        // 16384 rows total

// Padded pitches (f16 elements) to spread LDS banks; all keep 16B alignment.
#define QP  1032             // Q tile pitch   (1024 + 8)
#define KP  72               // K/B tile pitch (64 + 8)
#define PP  72               // P tile pitch
#define VP  72               // Vt tile pitch

typedef _Float16 v16h __attribute__((ext_vector_type(16)));
typedef _Float16 v8h  __attribute__((ext_vector_type(8)));
typedef _Float16 v4h  __attribute__((ext_vector_type(4)));
typedef _Float16 v2h  __attribute__((ext_vector_type(2)));
typedef float    v8f  __attribute__((ext_vector_type(8)));

__device__ __forceinline__ v8f vzero8() {
  v8f z;
#pragma unroll
  for (int i = 0; i < 8; i++) z[i] = 0.0f;
  return z;
}

// A-fragment (16x32 f16): lane holds 16 contiguous K at k0+h*16.
__device__ __forceinline__ v16h ld_fragA(const _Float16* p) {
  v8h a = *(const v8h*)(p);
  v8h b = *(const v8h*)(p + 8);
  return __builtin_shufflevector(a, b, 0,1,2,3,4,5,6,7,8,9,10,11,12,13,14,15);
}

// B-fragment (32x16 f16): lane holds 16 contiguous K at k0+h*16.
__device__ __forceinline__ v16h ld_fragB(const _Float16* p) {
  v8h a = *(const v8h*)(p);
  v8h b = *(const v8h*)(p + 8);
  return __builtin_shufflevector(a, b, 0,1,2,3,4,5,6,7,8,9,10,11,12,13,14,15);
}

__device__ __forceinline__ v8f wmma16(v16h a, v16h b, v8f c) {
  return __builtin_amdgcn_wmma_f32_16x16x32_f16(
      false, a, false, b, (short)0, c, false, false);
}

// CDNA5 async copy: 16B per lane, global -> LDS, tracked by ASYNCcnt.
// GVS addressing: saddr(64b SGPR base) + vaddr(32b byte offset).
__device__ __forceinline__ void async_b128(void* lds, const void* gbase,
                                           unsigned goff) {
  unsigned l = (unsigned)(size_t)lds;
  asm volatile("global_load_async_to_lds_b128 %0, %1, %2"
               :: "v"(l), "v"(goff), "s"(gbase) : "memory");
}
__device__ __forceinline__ void wait_async_0()  { asm volatile("s_wait_asynccnt 0x0" ::: "memory"); }
__device__ __forceinline__ void wait_async_2()  { asm volatile("s_wait_asynccnt 0x2" ::: "memory"); }
__device__ __forceinline__ void wait_async_8()  { asm volatile("s_wait_asynccnt 0x8" ::: "memory"); }

// ---------------------------------------------------------------------------
// Kernel 0: prep. z=0: X fp32 -> f16.  z=1..3: W fp32 -> f16, transposed
// (Wt[n][k]) so the GEMM B-tile staging becomes a pure async row copy.
// ---------------------------------------------------------------------------
__global__ __launch_bounds__(256)
void prep(const float* __restrict__ X,
          const float* __restrict__ Wq, const float* __restrict__ Wk,
          const float* __restrict__ Wv,
          _Float16* __restrict__ Xh, _Float16* __restrict__ Wqt,
          _Float16* __restrict__ Wkt, _Float16* __restrict__ Wvt)
{
  const int z = blockIdx.z;
  const int idx = blockIdx.x * 256 + threadIdx.x;
  if (z == 0) {                      // 16M floats, 4 per thread
    size_t b = (size_t)idx * 4;
    float4 f = *(const float4*)(X + b);
    v4h hh; hh[0] = (_Float16)f.x; hh[1] = (_Float16)f.y;
            hh[2] = (_Float16)f.z; hh[3] = (_Float16)f.w;
    *(v4h*)(Xh + b) = hh;
  } else {
    if (idx >= 1024 * 1024) return;
    const float* src = (z == 1) ? Wq : (z == 2) ? Wk : Wv;
    _Float16*    dst = (z == 1) ? Wqt : (z == 2) ? Wkt : Wvt;
    int n = idx >> 10, k = idx & 1023;
    dst[(size_t)n * 1024 + k] = (_Float16)src[(size_t)k * 1024 + n];
  }
}

// ---------------------------------------------------------------------------
// Kernel 1: projection GEMM, 128x128 block tile, BK=64, double-buffered
// async staging. 8 waves, wave tile 32x64 -> 16 wmma per phase.
// ---------------------------------------------------------------------------
__global__ __launch_bounds__(256)
void proj_gemm(const _Float16* __restrict__ Xh,
               const _Float16* __restrict__ Wqt, const float* __restrict__ bq,
               const _Float16* __restrict__ Wkt, const float* __restrict__ bk,
               const _Float16* __restrict__ Wvt, const float* __restrict__ bv,
               _Float16* __restrict__ Qh, _Float16* __restrict__ Kh,
               _Float16* __restrict__ Vh)
{
  __shared__ __align__(16) _Float16 As[2][128 * KP];   // 2 x 18KB
  __shared__ __align__(16) _Float16 Bs[2][128 * KP];   // 2 x 18KB

  const _Float16* Wt; const float* bias; _Float16* Out;
  if (blockIdx.z == 0)      { Wt = Wqt; bias = bq; Out = Qh; }
  else if (blockIdx.z == 1) { Wt = Wkt; bias = bk; Out = Kh; }
  else                      { Wt = Wvt; bias = bv; Out = Vh; }

  const int tid  = threadIdx.x;
  const int lane = tid & 31;
  const int wave = tid >> 5;
  const int wm   = wave >> 1;          // 0..3 : 32-row strip
  const int wn   = wave & 1;           // 0..1 : 64-col strip
  const int r    = lane & 15;
  const int h    = (lane >> 4) & 1;
  const int m0   = blockIdx.y * 128;
  const int n0   = blockIdx.x * 128;

  const int srow = tid >> 3;           // staging: 8 x b128 per 64-wide row
  const int sc8  = tid & 7;

  v8f acc[2][4];
#pragma unroll
  for (int i = 0; i < 2; i++)
#pragma unroll
    for (int j = 0; j < 4; j++) acc[i][j] = vzero8();

  // Issue one 128x64 chunk of A and B (8 async ops / thread).
  auto issue = [&](int buf, int kk0) {
#pragma unroll
    for (int p = 0; p < 4; p++) {
      int row = srow + p * 32;
      unsigned ga = (unsigned)(((m0 + row) * 1024 + kk0 + sc8 * 8) * 2);
      async_b128(&As[buf][row * KP + sc8 * 8], Xh, ga);
      unsigned gb = (unsigned)(((n0 + row) * 1024 + kk0 + sc8 * 8) * 2);
      async_b128(&Bs[buf][row * KP + sc8 * 8], Wt, gb);
    }
  };

  issue(0, 0);
  for (int ph = 0; ph < 16; ph++) {
    const int cur = ph & 1;
    __syncthreads();                       // close reads of buf[cur^1]
    if (ph + 1 < 16) { issue(cur ^ 1, (ph + 1) * 64); wait_async_8(); }
    else             { wait_async_0(); }
    __syncthreads();                       // chunk ph visible to all
#pragma unroll
    for (int ks = 0; ks < 64; ks += 32) {
      v16h a0 = ld_fragA(&As[cur][(wm * 32 +  0 + r) * KP + ks + h * 16]);
      v16h a1 = ld_fragA(&As[cur][(wm * 32 + 16 + r) * KP + ks + h * 16]);
#pragma unroll
      for (int ct = 0; ct < 4; ct++) {
        v16h b = ld_fragB(&Bs[cur][(wn * 64 + ct * 16 + r) * KP + ks + h * 16]);
        acc[0][ct] = wmma16(a0, b, acc[0][ct]);
        acc[1][ct] = wmma16(a1, b, acc[1][ct]);
      }
    }
  }

  // Epilogue: +bias, f16 store.
#pragma unroll
  for (int rt = 0; rt < 2; rt++)
#pragma unroll
    for (int ct = 0; ct < 4; ct++) {
      int col = n0 + wn * 64 + ct * 16 + r;
      float bsc = bias[col];
#pragma unroll
      for (int i = 0; i < 8; i++) {
        int grow = m0 + wm * 32 + rt * 16 + i + h * 8;
        Out[(size_t)grow * 1024 + col] = (_Float16)(acc[rt][ct][i] + bsc);
      }
    }
}

// ---------------------------------------------------------------------------
// Kernel 2a: softmax stats (row max m, sum-exp l) over all 2048 keys.
// 64-query tile; Q resident in LDS (async-loaded, padded pitch); K streamed
// in 64x64 double-buffered async chunks.
// ---------------------------------------------------------------------------
__global__ __launch_bounds__(256)
void attn_stats(const _Float16* __restrict__ Qh, const _Float16* __restrict__ Kh,
                float* __restrict__ gm, float* __restrict__ gl)
{
  extern __shared__ char smem_a[];
  _Float16* Qs   = (_Float16*)smem_a;                       // 64*QP  = 132096B
  _Float16* Ks0  = (_Float16*)(smem_a + 132096);            // 64*KP  =   9216B
  _Float16* Ks1  = (_Float16*)(smem_a + 132096 + 9216);     //            9216B
  float*    sbuf = (float*)(smem_a + 150528);               // 64*65*4=  16640B
  float*    mrow = (float*)(smem_a + 167168);               // 64
  float*    lrow = (float*)(smem_a + 167424);               // 64
  _Float16* Ks[2] = { Ks0, Ks1 };

  const int tid  = threadIdx.x, lane = tid & 31, wave = tid >> 5;
  const int r    = lane & 15, h = (lane >> 4) & 1;
  const int q0   = blockIdx.x * 64;
  const size_t bS = (size_t)blockIdx.y * S_;
  const int rowb = (wave >> 1) * 16;
  const int colb = (wave & 1) * 32;
  const float scale = 0.03125f;          // 1/sqrt(1024)
  const int srow = tid >> 3, sc8 = tid & 7;

  // Async-load Q tile (64x1024 f16, 32 x b128 per thread).
#pragma unroll 4
  for (int p = 0; p < 32; p++) {
    int q = tid + p * 256;
    int row = q >> 7, c8 = q & 127;
    unsigned g = (unsigned)(((bS + q0 + row) * 1024 + c8 * 8) * 2);
    async_b128(Qs + row * QP + c8 * 8, Qh, g);
  }
  if (tid < 64) { mrow[tid] = -1e30f; lrow[tid] = 0.0f; }

  auto issueK = [&](_Float16* buf, int s0, int d0) {
#pragma unroll
    for (int p = 0; p < 2; p++) {
      int row = srow + p * 32;
      unsigned g = (unsigned)(((bS + s0 + row) * 1024 + d0 + sc8 * 8) * 2);
      async_b128(buf + row * KP + sc8 * 8, Kh, g);
    }
  };

  for (int s0 = 0; s0 < S_; s0 += 64) {
    issueK(Ks[0], s0, 0);
    v8f acc0 = vzero8(), acc1 = vzero8();
    for (int ph = 0; ph < 16; ph++) {
      const int cur = ph & 1;
      __syncthreads();
      if (ph + 1 < 16) { issueK(Ks[cur ^ 1], s0, (ph + 1) * 64); wait_async_2(); }
      else             { wait_async_0(); }
      __syncthreads();
#pragma unroll
      for (int ks = 0; ks < 64; ks += 32) {
        v16h a  = ld_fragA(Qs + (rowb + r) * QP + ph * 64 + ks + h * 16);
        v16h b0 = ld_fragB(Ks[cur] + (colb +  0 + r) * KP + ks + h * 16);
        v16h b1 = ld_fragB(Ks[cur] + (colb + 16 + r) * KP + ks + h * 16);
        acc0 = wmma16(a, b0, acc0);
        acc1 = wmma16(a, b1, acc1);
      }
    }
#pragma unroll
    for (int i = 0; i < 8; i++) {
      int rowL = rowb + i + h * 8;
      sbuf[rowL * 65 + colb +  0 + r] = acc0[i] * scale;
      sbuf[rowL * 65 + colb + 16 + r] = acc1[i] * scale;
    }
    __syncthreads();
    if (tid < 64) {                       // padded pitch: conflict-free
      float mo = mrow[tid];
      float mx = mo;
      const float* sr = sbuf + tid * 65;
      for (int c = 0; c < 64; c++) mx = fmaxf(mx, sr[c]);
      float sum = 0.0f;
      for (int c = 0; c < 64; c++) sum += __expf(sr[c] - mx);
      lrow[tid] = lrow[tid] * __expf(mo - mx) + sum;
      mrow[tid] = mx;
    }
    __syncthreads();
  }
  if (tid < 64) {
    gm[bS + q0 + tid] = mrow[tid];
    gl[bS + q0 + tid] = lrow[tid];
  }
}

// ---------------------------------------------------------------------------
// Kernel 2b: O = softmax(QK^T)·V. 32-query tile; wave w owns value columns
// [w*128, +128) (O accum = 128 VGPRs/lane). Scores recomputed per key block;
// V staged transposed per wave (pair-packed b32 stores + XOR-16 swizzle).
// ---------------------------------------------------------------------------
__global__ __launch_bounds__(256)
void attn_out(const _Float16* __restrict__ Qh, const _Float16* __restrict__ Kh,
              const _Float16* __restrict__ Vh,
              const float* __restrict__ gm, const float* __restrict__ gl,
              float* __restrict__ Out)
{
  extern __shared__ char smem_b[];
  _Float16* Qs   = (_Float16*)smem_b;                       // 32*QP = 66048B
  _Float16* Ks0  = (_Float16*)(smem_b + 66048);             // 64*KP =  9216B
  _Float16* Ks1  = (_Float16*)(smem_b + 66048 + 9216);      //          9216B
  _Float16* Ps   = (_Float16*)(smem_b + 84480);             // 32*PP =  4608B
  _Float16* Vt   = (_Float16*)(smem_b + 89088);             // 8*128*VP=147456B
  float*    mrow = (float*)(smem_b + 236544);               // 32
  float*    linv = (float*)(smem_b + 236672);               // 32
  _Float16* Ks[2] = { Ks0, Ks1 };

  const int tid  = threadIdx.x, lane = tid & 31, wave = tid >> 5;
  const int r    = lane & 15, h = (lane >> 4) & 1;
  const int q0   = blockIdx.x * 32;
  const size_t bS = (size_t)blockIdx.y * S_;
  const int vbase = wave * 128;
  _Float16* Vtw  = Vt + wave * (128 * VP);
  const int srowb = (wave & 1) * 16;
  const int scolb = (wave >> 1) * 16;
  const float scale = 0.03125f;
  const int srow = tid >> 3, sc8 = tid & 7;

#pragma unroll 4
  for (int p = 0; p < 16; p++) {          // async Q tile (32x1024)
    int q = tid + p * 256;
    int row = q >> 7, c8 = q & 127;
    unsigned g = (unsigned)(((bS + q0 + row) * 1024 + c8 * 8) * 2);
    async_b128(Qs + row * QP + c8 * 8, Qh, g);
  }
  if (tid < 32) {
    mrow[tid] = gm[bS + q0 + tid];
    linv[tid] = 1.0f / gl[bS + q0 + tid];
  }

  auto issueK = [&](_Float16* buf, int s0, int d0) {
#pragma unroll
    for (int p = 0; p < 2; p++) {
      int row = srow + p * 32;
      unsigned g = (unsigned)(((bS + s0 + row) * 1024 + d0 + sc8 * 8) * 2);
      async_b128(buf + row * KP + sc8 * 8, Kh, g);
    }
  };

  v8f oacc[2][8];
#pragma unroll
  for (int i = 0; i < 2; i++)
#pragma unroll
    for (int j = 0; j < 8; j++) oacc[i][j] = vzero8();

  for (int s0 = 0; s0 < S_; s0 += 64) {
    issueK(Ks[0], s0, 0);

    // Stage V chunk transposed+swizzled: Vtw[c][key ^ ((c>>3 & 3)<<4)].
    {
      const int c0 = r * 8;
      const int sw = (r & 3) << 4;
      for (int kk = 0; kk < 64; kk += 4) {
        int key0 = kk + h * 2;
        const _Float16* src = Vh + (bS + s0 + key0) * 1024 + vbase + c0;
        v8h va = *(const v8h*)(src);
        v8h vb = *(const v8h*)(src + 1024);
#pragma unroll
        for (int u = 0; u < 8; u++) {
          v2h pr; pr[0] = va[u]; pr[1] = vb[u];
          *(v2h*)(Vtw + (c0 + u) * VP + (key0 ^ sw)) = pr;
        }
      }
    }

    // Scores: 32x64 tile, one 16x16 WMMA tile per wave.
    v8f sc = vzero8();
    for (int ph = 0; ph < 16; ph++) {
      const int cur = ph & 1;
      __syncthreads();
      if (ph + 1 < 16) { issueK(Ks[cur ^ 1], s0, (ph + 1) * 64); wait_async_2(); }
      else             { wait_async_0(); }
      __syncthreads();
#pragma unroll
      for (int ks = 0; ks < 64; ks += 32) {
        v16h a = ld_fragA(Qs + (srowb + r) * QP + ph * 64 + ks + h * 16);
        v16h b = ld_fragB(Ks[cur] + (scolb + r) * KP + ks + h * 16);
        sc = wmma16(a, b, sc);
      }
    }

    // P = exp(s*scale - m) / l   (normalization folded in).
#pragma unroll
    for (int i = 0; i < 8; i++) {
      int rowL = srowb + i + h * 8;
      float p = __expf(sc[i] * scale - mrow[rowL]) * linv[rowL];
      Ps[rowL * PP + scolb + r] = (_Float16)p;
    }
    __syncthreads();

    // O += P(32x64) * V(64x128).
#pragma unroll
    for (int ks = 0; ks < 64; ks += 32) {
      v16h a0 = ld_fragA(Ps + ( 0 + r) * PP + ks + h * 16);
      v16h a1 = ld_fragA(Ps + (16 + r) * PP + ks + h * 16);
#pragma unroll
      for (int ct = 0; ct < 8; ct++) {
        int c = ct * 16 + r;
        int sw2 = ((c >> 3) & 3) << 4;
        v16h b = ld_fragB(Vtw + c * VP + ((ks + h * 16) ^ sw2));
        oacc[0][ct] = wmma16(a0, b, oacc[0][ct]);
        oacc[1][ct] = wmma16(a1, b, oacc[1][ct]);
      }
    }
    __syncthreads();
  }

  // fp32 output.
#pragma unroll
  for (int rt = 0; rt < 2; rt++)
#pragma unroll
    for (int ct = 0; ct < 8; ct++) {
      int col = vbase + ct * 16 + r;
#pragma unroll
      for (int i = 0; i < 8; i++) {
        int row = q0 + rt * 16 + i + h * 8;
        Out[(bS + row) * 1024 + col] = oacc[rt][ct][i];
      }
    }
}

// ---------------------------------------------------------------------------
extern "C" void kernel_launch(void* const* d_in, const int* in_sizes, int n_in,
                              void* d_out, int out_size, void* d_ws, size_t ws_size,
                              hipStream_t stream) {
  (void)in_sizes; (void)n_in; (void)out_size; (void)ws_size;
  const float* X  = (const float*)d_in[0];
  const float* Wq = (const float*)d_in[1];
  const float* bq = (const float*)d_in[2];
  const float* Wk = (const float*)d_in[3];
  const float* bk = (const float*)d_in[4];
  const float* Wv = (const float*)d_in[5];
  const float* bv = (const float*)d_in[6];
  float* Out = (float*)d_out;

  // Workspace (~141 MB): Qh/Kh/Vh/Xh f16 + Wt f16 x3 + m,l f32.
  char* ws = (char*)d_ws;
  const size_t mat = (size_t)M_ * D_ * sizeof(_Float16);   // 32 MB
  const size_t wsz = (size_t)D_ * 1024 * sizeof(_Float16); //  2 MB
  _Float16* Qh  = (_Float16*)(ws);
  _Float16* Kh  = (_Float16*)(ws + mat);
  _Float16* Vh  = (_Float16*)(ws + 2 * mat);
  _Float16* Xh  = (_Float16*)(ws + 3 * mat);
  _Float16* Wqt = (_Float16*)(ws + 4 * mat);
  _Float16* Wkt = (_Float16*)(ws + 4 * mat + wsz);
  _Float16* Wvt = (_Float16*)(ws + 4 * mat + 2 * wsz);
  float*    gm  = (float*)(ws + 4 * mat + 3 * wsz);
  float*    gl  = gm + M_;

  prep<<<dim3(16384, 1, 4), 256, 0, stream>>>(X, Wq, Wk, Wv, Xh, Wqt, Wkt, Wvt);

  proj_gemm<<<dim3(1024 / 128, M_ / 128, 3), 256, 0, stream>>>(
      Xh, Wqt, bq, Wkt, bk, Wvt, bv, Qh, Kh, Vh);

  attn_stats<<<dim3(S_ / 64, B_), 256, 167680, stream>>>(Qh, Kh, gm, gl);

  attn_out<<<dim3(S_ / 32, B_), 256, 236800, stream>>>(Qh, Kh, Vh, gm, gl, Out);
}
